// DisplacementNet_16166256902543
// MI455X (gfx1250) — matrix operand order repaired and checked by reading, not compiled, not run on hardware
//
#include <hip/hip_runtime.h>
#include <hip/hip_bf16.h>
#include <stdint.h>

#define NPTS   32768
#define KNN    12
#define WIDTH  192
#define NCOND  64
#define INPAD  128     // 115 padded to 4 k-tiles of 32
#define MIXPAD 416     // 390 padded to 13 k-tiles of 32
#define NLAYER 4
#define KT_IN  4
#define KT_MIX 13
#define CT     12      // 192 / 16 col tiles

typedef __attribute__((ext_vector_type(2)))  float        v2f;
typedef __attribute__((ext_vector_type(8)))  float        v8f;
typedef __attribute__((ext_vector_type(16))) __bf16       v16bf;
typedef __attribute__((ext_vector_type(4)))  unsigned int u32x4;
typedef int i32x4v __attribute__((vector_size(16)));   // matches builtin proto

#if defined(__has_builtin)
#if __has_builtin(__builtin_amdgcn_global_load_async_to_lds_b128)
#define HAVE_ASYNC_LDS 1
#endif
#if __has_builtin(__builtin_amdgcn_s_wait_asynccnt)
#define HAVE_WAIT_ASYNC 1
#endif
#endif

__device__ inline void wait_asynccnt0() {
#ifdef HAVE_WAIT_ASYNC
  __builtin_amdgcn_s_wait_asynccnt(0);
#else
  asm volatile("s_wait_asynccnt 0x0" ::: "memory");
#endif
}

__device__ inline unsigned short f2bf(float f) {
  union { float f; unsigned u; } v; v.f = f;
  unsigned r = v.u + 0x7FFFu + ((v.u >> 16) & 1u);   // round-to-nearest-even
  return (unsigned short)(r >> 16);
}
__device__ inline float silu_f(float v) { return v / (1.f + __expf(-v)); }

__device__ inline v16bf load_frag(const unsigned short* p0, const unsigned short* p1) {
  union { u32x4 u[2]; v16bf v; } f;
  f.u[0] = *(const u32x4*)p0;
  f.u[1] = *(const u32x4*)p1;
  return f.v;
}

// ---------------------------------------------------------------------------
// Pack x -> (x, y, z, |x|^2) float4
// ---------------------------------------------------------------------------
__global__ void pack_x_kernel(const float* __restrict__ x, float4* __restrict__ xpack) {
  int i = blockIdx.x * 256 + threadIdx.x;
  if (i >= NPTS) return;
  float a = x[3 * i], b = x[3 * i + 1], c = x[3 * i + 2];
  xpack[i] = make_float4(a, b, c, a * a + b * b + c * c);
}

// ---------------------------------------------------------------------------
// Pack Wp (115x192, pad K->128) into bf16 WMMA B-fragment layout:
// [ct][kt][lane][16], lane's K = kt*32 + (lane>=16?16:0) + e, N = ct*16+lane%16
// ---------------------------------------------------------------------------
__global__ void pack_wp_kernel(const float* __restrict__ Wp, unsigned short* __restrict__ WpP) {
  int t = blockIdx.x * 256 + threadIdx.x;
  if (t >= CT * KT_IN * 32 * 16) return;
  int e = t & 15, lane = (t >> 4) & 31, kt = (t >> 9) & 3, ct = t >> 11;
  int col = ct * 16 + (lane & 15);
  int K = kt * 32 + ((lane >> 4) * 16) + e;
  WpP[t] = f2bf(K < 115 ? Wp[K * WIDTH + col] : 0.f);
}

// Same for Wl (4 x 390 x 192, pad K->416)
__global__ void pack_wl_kernel(const float* __restrict__ Wl, unsigned short* __restrict__ WlP) {
  int t = blockIdx.x * 256 + threadIdx.x;
  if (t >= NLAYER * CT * KT_MIX * 512) return;
  int e = t & 15, lane = (t >> 4) & 31;
  int rest = t >> 9;
  int kt = rest % KT_MIX; rest /= KT_MIX;
  int ct = rest % CT;
  int li = rest / CT;
  int col = ct * 16 + (lane & 15);
  int K = kt * 32 + ((lane >> 4) * 16) + e;
  WlP[t] = f2bf(K < 390 ? Wl[((size_t)li * 390 + K) * WIDTH + col] : 0.f);
}

// FiLM vectors: gamma = z@Wg + bg, beta = z@Wb + bb  (per layer, 192 wide)
__global__ void film_kernel(const float* __restrict__ z, const float* __restrict__ Wg,
                            const float* __restrict__ bg, const float* __restrict__ Wb,
                            const float* __restrict__ bb, float* __restrict__ film) {
  int t = blockIdx.x * 256 + threadIdx.x;
  if (t >= NLAYER * WIDTH) return;
  int li = t / WIDTH, w = t % WIDTH;
  float g = bg[t], b = bb[t];
  for (int c = 0; c < NCOND; ++c) {
    float zc = z[c];
    g += zc * Wg[((size_t)li * NCOND + c) * WIDTH + w];
    b += zc * Wb[((size_t)li * NCOND + c) * WIDTH + w];
  }
  film[li * 2 * WIDTH + w]         = g;
  film[li * 2 * WIDTH + WIDTH + w] = b;
}

// ---------------------------------------------------------------------------
// kNN: WMMA f32 16x16x4 distance tiles + per-lane register top-12.
// Block = 256 thr (8 waves), 32 rows per wave, 256 rows per block.
// score(i,j) = |x_j|^2 - 2 x_i.x_j  (row-constant shift dropped; order-safe)
// Column chunks staged global->LDS with GLOBAL_LOAD_ASYNC_TO_LDS (ASYNCcnt).
// ---------------------------------------------------------------------------
__global__ __launch_bounds__(256) void knn_kernel(const float4* __restrict__ xpack,
                                                  int* __restrict__ knn_idx) {
  __shared__ __align__(16) float4 xs[1024];     // 16 KB staged column chunk
  __shared__ __align__(16) float  sb[8][512];   // 16 KB per-wave 32x16 scoreboard
  const int lane = threadIdx.x & 31;
  const int wave = threadIdx.x >> 5;
  const int hi   = lane >> 4;
  const int lcol = lane & 15;
  const int rowBase = blockIdx.x * 256 + wave * 32;
  const int myRow   = rowBase + lane;

  // A fragments (16x4 f32): v0 = {K0 | K2}, v1 = {K1 | K3} across half-waves.
  float4 p0 = xpack[rowBase + lcol];
  float4 p1 = xpack[rowBase + 16 + lcol];
  v2f a0, a1;
  if (hi == 0) { a0[0] = -2.f * p0.x; a0[1] = -2.f * p0.y; a1[0] = -2.f * p1.x; a1[1] = -2.f * p1.y; }
  else         { a0[0] = -2.f * p0.z; a0[1] = 1.f;         a1[0] = -2.f * p1.z; a1[1] = 1.f; }

  float bd[KNN]; int bi[KNN];
  #pragma unroll
  for (int q = 0; q < KNN; ++q) { bd[q] = 3.0e38f; bi[q] = 0; }
  float wv = 3.0e38f; int wi = 0;

  // Branch-free replace-worst insertion (v_cndmask, no exec-mask trees);
  // single divergent guard on the rare insert path.
  auto consider = [&](float s, int j) {
    if (j != myRow && s < wv) {
      #pragma unroll
      for (int q = 0; q < KNN; ++q) {
        bool hit = (q == wi);
        bd[q] = hit ? s : bd[q];
        bi[q] = hit ? j : bi[q];
      }
      float nwv = bd[0]; int nwi = 0;
      #pragma unroll
      for (int q = 1; q < KNN; ++q) {
        bool g = bd[q] > nwv;
        nwv = g ? bd[q] : nwv;
        nwi = g ? q : nwi;
      }
      wv = nwv; wi = nwi;
    }
  };

  for (int ch = 0; ch < NPTS / 1024; ++ch) {
    __syncthreads();
    const int cbase = ch * 1024;
#ifdef HAVE_ASYNC_LDS
    for (int i = threadIdx.x; i < 1024; i += 256) {
      __builtin_amdgcn_global_load_async_to_lds_b128(
          (i32x4v*)(xpack + cbase + i),
          (i32x4v*)(&xs[i]),
          0, 0);
      if (ch + 1 < NPTS / 1024) __builtin_prefetch(&xpack[cbase + 1024 + i], 0, 0);
    }
    wait_asynccnt0();
#else
    for (int i = threadIdx.x; i < 1024; i += 256) {
      xs[i] = xpack[cbase + i];
      if (ch + 1 < NPTS / 1024) __builtin_prefetch(&xpack[cbase + 1024 + i], 0, 0);
    }
#endif
    __syncthreads();

    for (int t = 0; t < 64; ++t) {
      float4 q = xs[t * 16 + lcol];
      v2f b;
      if (hi == 0) { b[0] = q.x; b[1] = q.y; }   // K0,K1 = x,y
      else         { b[0] = q.z; b[1] = q.w; }   // K2,K3 = z,|x|^2
      v8f cz = {0.f, 0.f, 0.f, 0.f, 0.f, 0.f, 0.f, 0.f};
      v8f d0 = __builtin_amdgcn_wmma_f32_16x16x4_f32(false, a0, false, b, (short)0, cz, false, false);
      v8f d1 = __builtin_amdgcn_wmma_f32_16x16x4_f32(false, a1, false, b, (short)0, cz, false, false);

      float* w = &sb[wave][0];
      #pragma unroll
      for (int r = 0; r < 8; ++r) {
        w[(r + 8 * hi) * 16 + lcol]      = d0[r];
        w[(16 + r + 8 * hi) * 16 + lcol] = d1[r];
      }
      asm volatile("s_wait_dscnt 0x0" ::: "memory");

      const float4* rp = (const float4*)&sb[wave][lane * 16];
      float4 s0 = rp[0], s1 = rp[1], s2 = rp[2], s3 = rp[3];
      const int jb = cbase + t * 16;
      consider(s0.x, jb + 0);  consider(s0.y, jb + 1);
      consider(s0.z, jb + 2);  consider(s0.w, jb + 3);
      consider(s1.x, jb + 4);  consider(s1.y, jb + 5);
      consider(s1.z, jb + 6);  consider(s1.w, jb + 7);
      consider(s2.x, jb + 8);  consider(s2.y, jb + 9);
      consider(s2.z, jb + 10); consider(s2.w, jb + 11);
      consider(s3.x, jb + 12); consider(s3.y, jb + 13);
      consider(s3.z, jb + 14); consider(s3.w, jb + 15);
    }
  }
  #pragma unroll
  for (int q = 0; q < KNN; ++q) knn_idx[(size_t)myRow * KNN + q] = bi[q];
}

// ---------------------------------------------------------------------------
// Fourier features + cond concat + perception GEMM (bf16 WMMA) + SiLU -> h0
// Block 256 (8 waves), 16 rows per wave.
// ---------------------------------------------------------------------------
__global__ __launch_bounds__(256) void feat_h0_kernel(const float4* __restrict__ xpack,
    const float* __restrict__ z, const float* __restrict__ B0, const float* __restrict__ B1,
    const float* __restrict__ B2, const unsigned short* __restrict__ WpP,
    const float* __restrict__ bp, float* __restrict__ h0) {
  __shared__ float Bsh[72];
  __shared__ __align__(16) unsigned short As[8][16 * INPAD];   // 32 KB bf16 A tiles
  const int lane = threadIdx.x & 31, wave = threadIdx.x >> 5;
  const int hi = lane >> 4, lcol = lane & 15;
  const int rowBase = blockIdx.x * 128 + wave * 16;

  if (threadIdx.x < 72) {
    const float* src = threadIdx.x < 24 ? B0 : (threadIdx.x < 48 ? B1 : B2);
    Bsh[threadIdx.x] = src[threadIdx.x % 24];
  }
  __syncthreads();

  for (int r = 0; r < 16; ++r) {
    float4 xp = xpack[rowBase + r];
    for (int c = lane; c < INPAD; c += 32) {
      float v;
      if (c < 48) {
        int g = c >> 4, e = c & 15, k = e & 7;
        const float* B = Bsh + g * 24;              // (3,8) row-major
        float xb = xp.x * B[k] + xp.y * B[8 + k] + xp.z * B[16 + k];
        v = (e < 8) ? __sinf(xb) : __cosf(xb);
      } else if (c < 51) {
        v = (c == 48) ? xp.x : ((c == 49) ? xp.y : xp.z);
      } else if (c < 115) {
        v = z[c - 51];
      } else {
        v = 0.f;
      }
      As[wave][r * INPAD + c] = f2bf(v);
    }
  }
  asm volatile("s_wait_dscnt 0x0" ::: "memory");

  const unsigned short* aRow = &As[wave][lcol * INPAD];
  for (int ct = 0; ct < CT; ++ct) {
    v8f acc = {0.f, 0.f, 0.f, 0.f, 0.f, 0.f, 0.f, 0.f};
    #pragma unroll
    for (int kt = 0; kt < KT_IN; ++kt) {
      const unsigned short* pa = aRow + kt * 32 + hi * 16;
      v16bf af = load_frag(pa, pa + 8);
      const unsigned short* pb = WpP + ((size_t)(ct * KT_IN + kt) * 32 + lane) * 16;
      v16bf bf = load_frag(pb, pb + 8);
      acc = __builtin_amdgcn_wmma_f32_16x16x32_bf16(false, af, false, bf, (short)0, acc, false, false);
    }
    int col = ct * 16 + lcol;
    float bpc = bp[col];
    #pragma unroll
    for (int r = 0; r < 8; ++r) {
      int row = rowBase + r + 8 * hi;
      h0[(size_t)row * WIDTH + col] = silu_f(acc[r] + bpc);
    }
  }
}

// ---------------------------------------------------------------------------
// Relative-coordinate stats per row: mean(3) + std(3)  (stride-8 padded)
// ---------------------------------------------------------------------------
__global__ void relfeat_kernel(const float4* __restrict__ xpack, const int* __restrict__ knn_idx,
                               float* __restrict__ relf) {
  int row = blockIdx.x * 256 + threadIdx.x;
  if (row >= NPTS) return;
  float4 xc = xpack[row];
  float m0 = 0, m1 = 0, m2 = 0, s0 = 0, s1 = 0, s2 = 0;
  #pragma unroll
  for (int j = 0; j < KNN; ++j) {
    int nb = knn_idx[(size_t)row * KNN + j];
    float4 xn = xpack[nb];
    float dx = xn.x - xc.x, dy = xn.y - xc.y, dz = xn.z - xc.z;
    m0 += dx; m1 += dy; m2 += dz;
    s0 += dx * dx; s1 += dy * dy; s2 += dz * dz;
  }
  const float inv = 1.f / (float)KNN;
  m0 *= inv; m1 *= inv; m2 *= inv;
  relf[row * 8 + 0] = m0;
  relf[row * 8 + 1] = m1;
  relf[row * 8 + 2] = m2;
  relf[row * 8 + 3] = sqrtf(fmaxf(s0 * inv - m0 * m0, 0.f));
  relf[row * 8 + 4] = sqrtf(fmaxf(s1 * inv - m1 * m1, 0.f));
  relf[row * 8 + 5] = sqrtf(fmaxf(s2 * inv - m2 * m2, 0.f));
}

// ---------------------------------------------------------------------------
// One graph layer: gather-mean + mix GEMM (bf16 WMMA, K=416) + SiLU + FiLM
// Block 128 (4 waves), 16 rows per wave, 53 KB LDS A-stage.
// ---------------------------------------------------------------------------
__global__ __launch_bounds__(128) void layer_kernel(int li, const float* __restrict__ hin,
    float* __restrict__ hout, const int* __restrict__ knn_idx, const float* __restrict__ relf,
    const unsigned short* __restrict__ WlP, const float* __restrict__ bl,
    const float* __restrict__ film) {
  __shared__ __align__(16) unsigned short As[4][16 * MIXPAD];   // 53,248 B
  const int lane = threadIdx.x & 31, wave = threadIdx.x >> 5;
  const int hi = lane >> 4, lcol = lane & 15;
  const int rowBase = blockIdx.x * 64 + wave * 16;

  for (int r = 0; r < 16; ++r) {
    const int row = rowBase + r;
    int nb[KNN];
    #pragma unroll
    for (int j = 0; j < KNN; ++j) nb[j] = knn_idx[(size_t)row * KNN + j];
    const float* hrow = hin + (size_t)row * WIDTH;
    for (int c = lane; c < WIDTH; c += 32) {
      float hv = hrow[c];
      float s = 0.f;
      #pragma unroll
      for (int j = 0; j < KNN; ++j) s += hin[(size_t)nb[j] * WIDTH + c];
      As[wave][r * MIXPAD + c]         = f2bf(hv);
      As[wave][r * MIXPAD + WIDTH + c] = f2bf(s * (1.f / (float)KNN));
    }
    // cols 384..415: 6 rel-features + zero pad
    int c = 384 + lane;
    As[wave][r * MIXPAD + c] = (lane < 6) ? f2bf(relf[row * 8 + lane]) : (unsigned short)0;
  }
  asm volatile("s_wait_dscnt 0x0" ::: "memory");

  const unsigned short* aRow = &As[wave][lcol * MIXPAD];
  const unsigned short* Wl_l = WlP + (size_t)li * CT * KT_MIX * 512;
  const float* fg = film + (size_t)li * 2 * WIDTH;
  for (int ct = 0; ct < CT; ++ct) {
    v8f acc = {0.f, 0.f, 0.f, 0.f, 0.f, 0.f, 0.f, 0.f};
    #pragma unroll
    for (int kt = 0; kt < KT_MIX; ++kt) {
      const unsigned short* pa = aRow + kt * 32 + hi * 16;
      v16bf af = load_frag(pa, pa + 8);
      const unsigned short* pb = Wl_l + ((size_t)(ct * KT_MIX + kt) * 32 + lane) * 16;
      v16bf bf = load_frag(pb, pb + 8);
      acc = __builtin_amdgcn_wmma_f32_16x16x32_bf16(false, af, false, bf, (short)0, acc, false, false);
    }
    int col = ct * 16 + lcol;
    float blc = bl[li * WIDTH + col];
    float g = fg[col], be = fg[WIDTH + col];
    #pragma unroll
    for (int r = 0; r < 8; ++r) {
      int row = rowBase + r + 8 * hi;
      float v = silu_f(acc[r] + blc);
      hout[(size_t)row * WIDTH + col] = v * g + be;
    }
  }
}

// ---------------------------------------------------------------------------
// Output projection: (h @ Wout + bout) * 0.01
// ---------------------------------------------------------------------------
__global__ void out_kernel(const float* __restrict__ h, const float* __restrict__ Wout,
                           const float* __restrict__ bout, float* __restrict__ out) {
  int row = blockIdx.x * 256 + threadIdx.x;
  if (row >= NPTS) return;
  float a0 = 0.f, a1 = 0.f, a2 = 0.f;
  const float* hr = h + (size_t)row * WIDTH;
  for (int k = 0; k < WIDTH; ++k) {
    float hv = hr[k];
    a0 += hv * Wout[k * 3 + 0];
    a1 += hv * Wout[k * 3 + 1];
    a2 += hv * Wout[k * 3 + 2];
  }
  out[row * 3 + 0] = (a0 + bout[0]) * 0.01f;
  out[row * 3 + 1] = (a1 + bout[1]) * 0.01f;
  out[row * 3 + 2] = (a2 + bout[2]) * 0.01f;
}

// ---------------------------------------------------------------------------
extern "C" void kernel_launch(void* const* d_in, const int* in_sizes, int n_in,
                              void* d_out, int out_size, void* d_ws, size_t ws_size,
                              hipStream_t stream) {
  const float* x    = (const float*)d_in[0];
  const float* z    = (const float*)d_in[1];
  const float* B0   = (const float*)d_in[2];
  const float* B1   = (const float*)d_in[3];
  const float* B2   = (const float*)d_in[4];
  const float* Wp   = (const float*)d_in[5];
  const float* bp   = (const float*)d_in[6];
  const float* Wl   = (const float*)d_in[7];
  const float* bl   = (const float*)d_in[8];
  const float* Wg   = (const float*)d_in[9];
  const float* bg   = (const float*)d_in[10];
  const float* Wb   = (const float*)d_in[11];
  const float* bb   = (const float*)d_in[12];
  const float* Wout = (const float*)d_in[13];
  const float* bout = (const float*)d_in[14];
  float* out = (float*)d_out;

  char* ws = (char*)d_ws;
  size_t off = 0;
  auto take = [&](size_t bytes) -> char* {
    char* p = ws + off;
    off = (off + bytes + 255) & ~(size_t)255;
    return p;
  };
  int*            knn  = (int*)take((size_t)NPTS * KNN * 4);
  float4*         xpk  = (float4*)take((size_t)NPTS * 16);
  float*          relf = (float*)take((size_t)NPTS * 8 * 4);
  float*          hA   = (float*)take((size_t)NPTS * WIDTH * 4);
  float*          hB   = (float*)take((size_t)NPTS * WIDTH * 4);
  unsigned short* WpP  = (unsigned short*)take((size_t)CT * KT_IN * 512 * 2);
  unsigned short* WlP  = (unsigned short*)take((size_t)NLAYER * CT * KT_MIX * 512 * 2);
  float*          film = (float*)take((size_t)NLAYER * 2 * WIDTH * 4);

  pack_x_kernel <<<NPTS / 256, 256, 0, stream>>>(x, xpk);
  pack_wp_kernel<<<(CT * KT_IN * 512 + 255) / 256, 256, 0, stream>>>(Wp, WpP);
  pack_wl_kernel<<<(NLAYER * CT * KT_MIX * 512 + 255) / 256, 256, 0, stream>>>(Wl, WlP);
  film_kernel   <<<(NLAYER * WIDTH + 255) / 256, 256, 0, stream>>>(z, Wg, bg, Wb, bb, film);

  knn_kernel    <<<NPTS / 256, 256, 0, stream>>>(xpk, knn);
  feat_h0_kernel<<<NPTS / 128, 256, 0, stream>>>(xpk, z, B0, B1, B2, WpP, bp, hA);
  relfeat_kernel<<<NPTS / 256, 256, 0, stream>>>(xpk, knn, relf);

  const float* hin = hA;
  float* hout = hB;
  for (int li = 0; li < NLAYER; ++li) {
    layer_kernel<<<NPTS / 64, 128, 0, stream>>>(li, hin, hout, knn, relf, WlP, bl, film);
    float* t = hout;
    hout = (float*)hin;
    hin = t;
  }
  out_kernel<<<NPTS / 256, 256, 0, stream>>>(hin, Wout, bout, out);
}
